// SplineCNNFPModule_68410239091221
// MI455X (gfx1250) — matrix operand order repaired and checked by reading, not compiled, not run on hardware
//
#include <hip/hip_runtime.h>
#include <hip/hip_bf16.h>

// ---------------- problem constants (from reference setup_inputs) ----------------
#define NC    8192      // coarse nodes
#define NF    16384     // fine nodes
#define NE    262144    // edges
#define CX    64        // x channels (coarse) == x_skip channels
#define CIN0  128       // layer-0 input channels (concat)
#define HCH   64        // hidden/out channels (all layers)
#define KT    126       // 125 spline kernels + 1 root slot

typedef unsigned short ushort_t;
typedef __attribute__((ext_vector_type(16))) __bf16 v16bf;
typedef __attribute__((ext_vector_type(8)))  float  v8f;

union Frag16 { v16bf v; uint4 q[2]; };

// LDS (addrspace 3) byte offset of a __shared__ pointer, for async-to-LDS dst
typedef __attribute__((address_space(3))) const void lds_cvoid;
__device__ __forceinline__ unsigned lds_offset_of(const void* p) {
  return (unsigned)(size_t)(lds_cvoid*)p;
}

// ---------------- bf16 helpers (RNE) ----------------
__device__ __forceinline__ ushort_t f2bf(float f) {
  union { float f; unsigned int u; } c; c.f = f;
  unsigned int r = c.u + 0x7FFFu + ((c.u >> 16) & 1u);
  return (ushort_t)(r >> 16);
}
__device__ __forceinline__ float bf2f(ushort_t u) {
  union { unsigned int u; float f; } c; c.u = ((unsigned int)u) << 16;
  return c.f;
}

// ---------------- 1) brute-force KNN(k=3) + inverse-d2 interpolate + concat ----------------
__global__ void __launch_bounds__(256) knn_interp_kernel(
    const float* __restrict__ x, const float* __restrict__ pos,
    const float* __restrict__ x_skip, const float* __restrict__ pos_skip,
    ushort_t* __restrict__ xbf)
{
  __shared__ float sp[256 * 3];
  const int i = blockIdx.x * 256 + threadIdx.x;      // NF multiple of 256
  const float py0 = pos_skip[i*3+0], py1 = pos_skip[i*3+1], py2 = pos_skip[i*3+2];
  float bd0 = 3.4e38f, bd1 = 3.4e38f, bd2 = 3.4e38f;
  int   bi0 = 0, bi1 = 0, bi2 = 0;
  for (int t = 0; t < NC / 256; ++t) {
    const int j = t * 256 + threadIdx.x;
    sp[threadIdx.x*3+0] = pos[j*3+0];
    sp[threadIdx.x*3+1] = pos[j*3+1];
    sp[threadIdx.x*3+2] = pos[j*3+2];
    __syncthreads();
    for (int jj = 0; jj < 256; ++jj) {
      const float dx = py0 - sp[jj*3+0];
      const float dy = py1 - sp[jj*3+1];
      const float dz = py2 - sp[jj*3+2];
      const float d2 = dx*dx + dy*dy + dz*dz;
      const int idx = t * 256 + jj;
      if (d2 < bd2) {                                // strict < keeps lowest index on ties
        if (d2 < bd0)      { bd2=bd1; bi2=bi1; bd1=bd0; bi1=bi0; bd0=d2; bi0=idx; }
        else if (d2 < bd1) { bd2=bd1; bi2=bi1; bd1=d2; bi1=idx; }
        else               { bd2=d2; bi2=idx; }
      }
    }
    __syncthreads();
  }
  const float w0 = 1.f / fmaxf(bd0, 1e-16f);
  const float w1 = 1.f / fmaxf(bd1, 1e-16f);
  const float w2 = 1.f / fmaxf(bd2, 1e-16f);
  const float inv = 1.f / (w0 + w1 + w2);
  const float* xa = x + (size_t)bi0 * CX;
  const float* xb = x + (size_t)bi1 * CX;
  const float* xc = x + (size_t)bi2 * CX;
  for (int c = 0; c < CX; ++c) {
    const float v = (w0 * xa[c] + w1 * xb[c] + w2 * xc[c]) * inv;
    xbf[(size_t)i * CIN0 + c] = f2bf(v);
  }
  for (int c = 0; c < CX; ++c)
    xbf[(size_t)i * CIN0 + CX + c] = f2bf(x_skip[(size_t)i * CX + c]);
}

// ---------------- 2) spline basis + kernel indices + dst degree counts ----------------
__global__ void __launch_bounds__(256) spline_basis_kernel(
    const int* __restrict__ ei, const float* __restrict__ pos_skip,
    float* __restrict__ basis, int* __restrict__ wi, int* __restrict__ cnt)
{
  const int e = blockIdx.x * 256 + threadIdx.x;
  if (e >= NE) return;
  const int src = ei[e], dst = ei[NE + e];
  float fr[3]; int lo[3];
#pragma unroll
  for (int d = 0; d < 3; ++d) {
    float p = (pos_skip[dst*3+d] - pos_skip[src*3+d]) * 4.0f + 0.5f; // /PATCH(0.25)
    p = fminf(fmaxf(p, 0.f), 1.f);
    const float v = p * 4.0f;                        // * (K_SIZE-1)
    const float flo = fminf(floorf(v), 3.0f);        // lo <= K-2
    fr[d] = v - flo;
    lo[d] = (int)flo;
  }
  const int off[3] = {1, 5, 25};
#pragma unroll
  for (int s = 0; s < 8; ++s) {
    float b = 1.f; int widx = 0;
#pragma unroll
    for (int d = 0; d < 3; ++d) {
      const int bit = (s >> d) & 1;
      b *= bit ? fr[d] : (1.f - fr[d]);
      widx += (lo[d] + bit) * off[d];
    }
    basis[(size_t)e*8 + s] = b;
    wi[(size_t)e*8 + s]    = widx;
  }
  atomicAdd(&cnt[dst], 1);
}

// ---------------- 3) cast W (+ root as slot 125) to bf16 ----------------
__global__ void __launch_bounds__(256) convert_weights_kernel(
    const float* __restrict__ W, const float* __restrict__ root,
    ushort_t* __restrict__ wc, int Cin)
{
  const int idx = blockIdx.x * 256 + threadIdx.x;
  const int per = Cin * HCH;
  if (idx >= KT * per) return;
  const int k = idx / per, r = idx - k * per;
  const float v = (k < 125) ? W[(size_t)k * per + r] : root[r];
  wc[idx] = f2bf(v);
}

// ---------------- 4) T[n,k,:] = X[n,:] @ Wk for all 126 k (bf16 WMMA, f32 acc) ----------
// One block per 16-row M-tile. X-tile async-staged to LDS once, A-fragments held in
// registers across all kernels handled by a wave; only B streams from L2.
// grid: (NF/16), block: 256 (8 waves); wave w handles kernels [16w, min(16w+16,126)).
template <int CIN>
__global__ void __launch_bounds__(256) gemm_xw_kernel(
    const ushort_t* __restrict__ X, const ushort_t* __restrict__ Wall,
    ushort_t* __restrict__ T)
{
  __shared__ __align__(16) ushort_t sA[16 * CIN];    // 16 x CIN bf16 tile (<= 4 KB)

  const int tid  = threadIdx.x;
  const int lane = tid & 31;
  const int wv   = tid >> 5;
  const int row0 = blockIdx.x * 16;

  // ---- async stage X tile into LDS (ASYNCcnt path), 8 bf16 per thread ----
  constexpr int chunks = (16 * CIN) / 8;             // b128 transfers
  if (tid < chunks) {
    const unsigned lds_off = lds_offset_of(&sA[tid * 8]);
    const unsigned long long gaddr =
        (unsigned long long)(size_t)(X + (size_t)row0 * CIN + tid * 8);
    asm volatile("global_load_async_to_lds_b128 %0, %1, off"
                 :: "v"(lds_off), "v"(gaddr) : "memory");
  }
  asm volatile("s_wait_asynccnt 0x0" ::: "memory");
  __syncthreads();

  // ---- A-fragments (16x32 bf16 layout): lane holds row m, K runs [g*8,+8) & [16+g*8,+8)
  const int m = lane & 15;
  const int g = lane >> 4;
  constexpr int NKK = CIN / 32;
  Frag16 afrag[NKK];
#pragma unroll
  for (int kk = 0; kk < NKK; ++kk) {
    const ushort_t* ar = &sA[m * CIN + kk * 32 + g * 8];
    afrag[kk].q[0] = *(const uint4*)ar;              // ds_load_b128
    afrag[kk].q[1] = *(const uint4*)(ar + 16);
  }

  const int kbeg = wv * 16;
  const int kend = (kbeg + 16 < KT) ? (kbeg + 16) : KT;
  for (int k = kbeg; k < kend; ++k) {
    const ushort_t* Wk = Wall + (size_t)k * CIN * HCH;
#pragma unroll
    for (int n0 = 0; n0 < HCH; n0 += 16) {
      v8f acc = {};
#pragma unroll
      for (int kk = 0; kk < NKK; ++kk) {
        Frag16 b;                                    // B 32x16: lane holds column n = m, K runs [g*8,+8) & [16+g*8,+8)
        const ushort_t* bcol = Wk + (size_t)(kk * 32 + g * 8) * HCH + n0 + m;
#pragma unroll
        for (int s = 0; s < 8; ++s) {
          ((ushort_t*)&b)[s]     = bcol[(size_t)s * HCH];
          ((ushort_t*)&b)[8 + s] = bcol[(size_t)(16 + s) * HCH];
        }
        acc = __builtin_amdgcn_wmma_f32_16x16x32_bf16(
            false, afrag[kk].v, false, b.v, (short)0, acc, false, false);
      }
      // D: lane holds column n = m, rows g*8 + j
      ushort_t* out = T + ((size_t)(row0 + g * 8) * KT + k) * HCH + n0 + m;
#pragma unroll
      for (int j = 0; j < 8; ++j)
        out[(size_t)j * KT * HCH] = f2bf(acc[j]);
    }
  }
}

// ---------------- 5) edge message: sum_s basis*T[src,wi_s,:]; atomic scatter to dst ----------------
__global__ void __launch_bounds__(256) msg_kernel(
    const ushort_t* __restrict__ T, const float* __restrict__ basis,
    const int* __restrict__ wi, const int* __restrict__ ei,
    float* __restrict__ agg)
{
  const int e = blockIdx.x * 8 + (threadIdx.x >> 5);
  const int lane = threadIdx.x & 31;
  if (e >= NE) return;
  const int src = ei[e], dst = ei[NE + e];
  float acc0 = 0.f, acc1 = 0.f;
#pragma unroll
  for (int s = 0; s < 8; ++s) {
    const float w = basis[(size_t)e*8 + s];
    const int   k = wi[(size_t)e*8 + s];
    const unsigned int* row =
        (const unsigned int*)(T + ((size_t)src * KT + k) * HCH);
    const unsigned int pr = row[lane];               // 2 packed bf16 channels
    acc0 += w * bf2f((ushort_t)(pr & 0xFFFFu));
    acc1 += w * bf2f((ushort_t)(pr >> 16));
  }
  unsafeAtomicAdd(&agg[(size_t)dst * HCH + 2*lane    ], acc0);
  unsafeAtomicAdd(&agg[(size_t)dst * HCH + 2*lane + 1], acc1);
}

// ---------------- 6) finalize: mean + root(T slot 125) + bias, ELU ----------------
__global__ void __launch_bounds__(256) finalize_kernel(
    const float* __restrict__ agg, const int* __restrict__ cnt,
    const ushort_t* __restrict__ T, const float* __restrict__ bias,
    ushort_t* __restrict__ xout_bf, float* __restrict__ out_f, int last)
{
  const int idx = blockIdx.x * 256 + threadIdx.x;    // NF*HCH
  const int n = idx >> 6, o = idx & 63;
  float v = agg[idx] / fmaxf((float)cnt[n], 1.f)
          + bf2f(T[((size_t)n * KT + 125) * HCH + o])
          + bias[o];
  v = (v > 0.f) ? v : expm1f(v);                     // ELU, alpha=1
  if (last) out_f[idx]   = v;
  else      xout_bf[idx] = f2bf(v);                  // next layer input, lda=64
}

// ---------------- host orchestration ----------------
extern "C" void kernel_launch(void* const* d_in, const int* in_sizes, int n_in,
                              void* d_out, int out_size, void* d_ws, size_t ws_size,
                              hipStream_t stream) {
  (void)in_sizes; (void)n_in; (void)out_size; (void)ws_size;
  const float* x        = (const float*)d_in[0];
  const float* pos      = (const float*)d_in[1];
  const float* x_skip   = (const float*)d_in[3];
  const float* pos_skip = (const float*)d_in[4];
  const int*   batch_sk = (const int*)d_in[5];
  const int*   edge_idx = (const int*)d_in[6];
  const float* Wl[3]    = {(const float*)d_in[7],  (const float*)d_in[10], (const float*)d_in[13]};
  const float* rootl[3] = {(const float*)d_in[8],  (const float*)d_in[11], (const float*)d_in[14]};
  const float* bl[3]    = {(const float*)d_in[9],  (const float*)d_in[12], (const float*)d_in[15]};

  // workspace carve-up (256B aligned slabs)
  char* w = (char*)d_ws;
  auto take = [&](size_t bytes) { char* p = w; w += (bytes + 255) & ~(size_t)255; return p; };
  float*    basis = (float*)   take((size_t)NE * 8 * 4);            // 8 MB
  int*      wi    = (int*)     take((size_t)NE * 8 * 4);            // 8 MB
  int*      cnt   = (int*)     take((size_t)NF * 4);                // 64 KB
  float*    agg   = (float*)   take((size_t)NF * HCH * 4);          // 4 MB
  ushort_t* xbf   = (ushort_t*)take((size_t)NF * CIN0 * 2);         // 4 MB
  ushort_t* wc    = (ushort_t*)take((size_t)KT * CIN0 * HCH * 2);   // 2 MB
  ushort_t* T     = (ushort_t*)take((size_t)NF * KT * HCH * 2);     // 252 MB

  hipMemsetAsync(cnt, 0, (size_t)NF * 4, stream);
  knn_interp_kernel<<<NF / 256, 256, 0, stream>>>(x, pos, x_skip, pos_skip, xbf);
  spline_basis_kernel<<<NE / 256, 256, 0, stream>>>(edge_idx, pos_skip, basis, wi, cnt);

  float* out_h = (float*)d_out;
  for (int layer = 0; layer < 3; ++layer) {
    const int Cin = (layer == 0) ? CIN0 : HCH;
    const int wtot = KT * Cin * HCH;
    convert_weights_kernel<<<(wtot + 255) / 256, 256, 0, stream>>>(
        Wl[layer], rootl[layer], wc, Cin);
    if (layer == 0)
      gemm_xw_kernel<CIN0><<<NF / 16, 256, 0, stream>>>(xbf, wc, T);
    else
      gemm_xw_kernel<HCH><<<NF / 16, 256, 0, stream>>>(xbf, wc, T);
    hipMemsetAsync(agg, 0, (size_t)NF * HCH * 4, stream);
    msg_kernel<<<NE / 8, 256, 0, stream>>>(T, basis, wi, edge_idx, agg);
    finalize_kernel<<<(NF * HCH) / 256, 256, 0, stream>>>(
        agg, cnt, T, bl[layer], xbf, out_h, layer == 2 ? 1 : 0);
  }

  // tuple tail: pos_skip (f32) then batch_skip (int bits, all zero)
  hipMemcpyAsync(out_h + (size_t)NF * HCH, pos_skip, (size_t)NF * 3 * 4,
                 hipMemcpyDeviceToDevice, stream);
  hipMemcpyAsync(out_h + (size_t)NF * (HCH + 3), batch_sk, (size_t)NF * 4,
                 hipMemcpyDeviceToDevice, stream);
}
